// GeneticOptimizationEngine_47004122088023
// MI455X (gfx1250) — compile-verified, matrix operand order checked by reading, not run on hardware
//
#include <hip/hip_runtime.h>
#include <cstdint>
#include <cstddef>

// ---------------------------------------------------------------------------
// GeneticOptimizationEngine for MI455X (gfx1250, wave32, WMMA)
//
// Pipeline (all on `stream`):
//   1) cvt_transpose_bf16 x3 : f32 weights -> bf16, transposed [n][k] in ws
//   2) fitness_kernel        : fused  relu(pop@W1+b1)@w2+b2 -> sigmoid
//                              (bf16 WMMA 16x16x32, f32 accumulate,
//                               32 rows/block, 2 accumulators/wave so each
//                               B fragment feeds two WMMAs)
//   3) topk_kernel           : 50 argmax sweeps over 65536 fitness values
//   4) mut1_kernel           : gelu(best@mutW1+b1) -> bf16 mid   (WMMA)
//   5) mut2_kernel           : mid@mutW2+b2, crossover blend -> out (WMMA)
// ---------------------------------------------------------------------------

typedef __attribute__((ext_vector_type(16))) __bf16 v16bf;
typedef __attribute__((ext_vector_type(8)))  float  v8f;

#define N_POP 65536
#define HDIM  2048
#define HHALF 1024
#define KSEL  50

// workspace layout (byte offsets, all 256B aligned)
#define WS_SELW1B  ((size_t)0)                         // 1024*2048 bf16, [n][k]
#define WS_MUTW1B  (WS_SELW1B + (size_t)HHALF*HDIM*2)  // 2048*2048 bf16, [n][k]
#define WS_MUTW2B  (WS_MUTW1B + (size_t)HDIM*HDIM*2)   // 2048*2048 bf16, [n][k]
#define WS_FITCOPY (WS_MUTW2B + (size_t)HDIM*HDIM*2)   // 65536 f32 (mutable copy)
#define WS_IDX     (WS_FITCOPY + (size_t)N_POP*4)      // 64 int
#define WS_MID     (WS_IDX + 256)                      // 64*2048 bf16

// --------------------------- weight conversion -----------------------------
// src: f32 row-major [K][Nn]  ->  dst: bf16 row-major [Nn][K]  (transposed)
__global__ void cvt_transpose_bf16(const float* __restrict__ src,
                                   __bf16* __restrict__ dst, int K, int Nn) {
    int total = K * Nn;
    for (int i = blockIdx.x * blockDim.x + threadIdx.x; i < total;
         i += gridDim.x * blockDim.x) {
        int k = i / Nn;
        int n = i - k * Nn;
        dst[(size_t)n * K + k] = (__bf16)src[i];
    }
}

// --------------------------- WMMA fragment loads ---------------------------
// A: 16 x HDIM bf16 tile in LDS, row-major. Lane m in 0..15 carries row m,
// lanes 16..31 carry the K+16 half.
__device__ __forceinline__ v16bf frag_a_lds(const __bf16* a, int k0, int lane) {
    int row = lane & 15;
    int kh  = lane >> 4;
    return *(const v16bf*)(a + row * HDIM + k0 + kh * 16);
}
// B: weights stored transposed [n][K=2048] bf16 in global -> contiguous 32B/lane
__device__ __forceinline__ v16bf frag_b_gbl(const __bf16* __restrict__ w,
                                            int n0, int k0, int lane) {
    int col = lane & 15;
    int kh  = lane >> 4;
    return *(const v16bf*)(w + (size_t)(n0 + col) * HDIM + k0 + kh * 16);
}

// ------------------------------ fitness ------------------------------------
// One block = 32 population rows (two 16-row A tiles, 128KB LDS). 8 waves,
// each wave owns 8 column tiles; per k-step one B fragment feeds WMMAs for
// both M tiles (2 accumulators). Epilogue fuses: +b1, relu, *w2[n], per-row
// LDS reduction, +b2, sigmoid.
__global__ void __launch_bounds__(256) fitness_kernel(
    const float* __restrict__ pop, const __bf16* __restrict__ w1b,
    const float* __restrict__ b1, const float* __restrict__ w2,
    const float* __restrict__ b2, float* __restrict__ fit_out,
    float* __restrict__ fit_copy) {
    extern __shared__ char smem[];
    __bf16* aTile  = (__bf16*)smem;                    // 32*2048 bf16 = 128KB
    float*  fitacc = (float*)(smem + 32 * HDIM * 2);   // 32 f32
    const int tid   = threadIdx.x;
    const int mbase = blockIdx.x * 32;

    if (tid < 32) fitacc[tid] = 0.0f;
    for (int e = tid; e < 32 * HDIM; e += 256) {
        int r = e >> 11, c = e & (HDIM - 1);
        aTile[e] = (__bf16)pop[(size_t)(mbase + r) * HDIM + c];
    }
    __syncthreads();

    const int wave = tid >> 5, lane = tid & 31;
    const __bf16* aT0 = aTile;               // rows  0..15
    const __bf16* aT1 = aTile + 16 * HDIM;   // rows 16..31
    for (int jt = 0; jt < 8; ++jt) {
        const int n0 = (wave * 8 + jt) * 16;
        v8f acc0 = {0.f, 0.f, 0.f, 0.f, 0.f, 0.f, 0.f, 0.f};
        v8f acc1 = {0.f, 0.f, 0.f, 0.f, 0.f, 0.f, 0.f, 0.f};
#pragma unroll 4
        for (int k0 = 0; k0 < HDIM; k0 += 32) {
            if (k0 + 32 < HDIM)  // prefetch next B fragment (global_prefetch_b8)
                __builtin_prefetch(w1b + (size_t)(n0 + (lane & 15)) * HDIM + k0 + 32, 0, 1);
            v16bf b  = frag_b_gbl(w1b, n0, k0, lane);
            v16bf a0 = frag_a_lds(aT0, k0, lane);
            v16bf a1 = frag_a_lds(aT1, k0, lane);
            acc0 = __builtin_amdgcn_wmma_f32_16x16x32_bf16(
                false, a0, false, b, (short)0, acc0, false, false);
            acc1 = __builtin_amdgcn_wmma_f32_16x16x32_bf16(
                false, a1, false, b, (short)0, acc1, false, false);
        }
        const int   n  = n0 + (lane & 15);
        const float wv = w2[n];
        const float bv = b1[n];
        const int   rb = (lane >> 4) * 8;  // lanes 0-15 -> rows 0..7, 16-31 -> 8..15
#pragma unroll
        for (int r = 0; r < 8; ++r) {
            float h0 = acc0[r] + bv;
            h0 = h0 > 0.f ? h0 : 0.f;
            atomicAdd(&fitacc[rb + r], h0 * wv);
            float h1 = acc1[r] + bv;
            h1 = h1 > 0.f ? h1 : 0.f;
            atomicAdd(&fitacc[16 + rb + r], h1 * wv);
        }
    }
    __syncthreads();
    if (tid < 32) {
        float x = fitacc[tid] + b2[0];
        float s = 1.0f / (1.0f + __expf(-x));
        fit_out[mbase + tid]  = s;
        fit_copy[mbase + tid] = s;
    }
}

// ------------------------------ top-k --------------------------------------
__global__ void __launch_bounds__(1024) topk_kernel(float* __restrict__ fit,
                                                    int* __restrict__ idx_out) {
    __shared__ float sval[1024];
    __shared__ int   sidx[1024];
    const int t = threadIdx.x;
    for (int iter = 0; iter < KSEL; ++iter) {
        float bv = -__builtin_huge_valf();
        int   bi = N_POP;
        for (int i = t; i < N_POP; i += 1024) {
            float v = fit[i];
            if (v > bv || (v == bv && i < bi)) { bv = v; bi = i; }
        }
        sval[t] = bv; sidx[t] = bi;
        __syncthreads();
        for (int s = 512; s > 0; s >>= 1) {
            if (t < s) {
                float ov = sval[t + s]; int oi = sidx[t + s];
                if (ov > sval[t] || (ov == sval[t] && oi < sidx[t])) {
                    sval[t] = ov; sidx[t] = oi;
                }
            }
            __syncthreads();
        }
        if (t == 0) { idx_out[iter] = sidx[0]; fit[sidx[0]] = -__builtin_huge_valf(); }
        __syncthreads();
    }
    if (t == 0)
        for (int i = KSEL; i < 64; ++i) idx_out[i] = 0;  // pad tiles
}

// ------------------------------ mutation 1 ---------------------------------
// grid (4 Mtiles, 16 col-groups); wave w owns one 16x16 output tile.
// mid = gelu(best @ mutW1 + b1) stored bf16 [64][2048].
__global__ void __launch_bounds__(256) mut1_kernel(
    const float* __restrict__ pop, const int* __restrict__ idx,
    const __bf16* __restrict__ w1b, const float* __restrict__ b1,
    __bf16* __restrict__ mid) {
    extern __shared__ char smem[];
    __bf16* aTile = (__bf16*)smem;                 // 16*2048 bf16
    int*    ridx  = (int*)(smem + 16 * HDIM * 2);  // 16 int
    const int tid = threadIdx.x;
    const int mt  = blockIdx.x;

    if (tid < 16) {
        int grow = mt * 16 + tid;
        ridx[tid] = (grow < KSEL) ? idx[grow] : 0;
    }
    __syncthreads();
    for (int e = tid; e < 16 * HDIM; e += 256) {
        int r = e >> 11, c = e & (HDIM - 1);
        aTile[e] = (__bf16)pop[(size_t)ridx[r] * HDIM + c];
    }
    __syncthreads();

    const int wave = tid >> 5, lane = tid & 31;
    const int n0   = (blockIdx.y * 8 + wave) * 16;
    v8f acc = {0.f, 0.f, 0.f, 0.f, 0.f, 0.f, 0.f, 0.f};
#pragma unroll 4
    for (int k0 = 0; k0 < HDIM; k0 += 32) {
        v16bf a = frag_a_lds(aTile, k0, lane);
        v16bf b = frag_b_gbl(w1b, n0, k0, lane);
        acc = __builtin_amdgcn_wmma_f32_16x16x32_bf16(
            false, a, false, b, (short)0, acc, false, false);
    }
    const int   n  = n0 + (lane & 15);
    const float bv = b1[n];
    const int   rb = (lane >> 4) * 8;
#pragma unroll
    for (int r = 0; r < 8; ++r) {
        float x = acc[r] + bv;
        float g = 0.5f * x * (1.0f + erff(x * 0.70710678118654752f));  // exact gelu
        int grow = mt * 16 + rb + r;
        mid[(size_t)grow * HDIM + n] = (__bf16)g;
    }
}

// ------------------------------ mutation 2 + crossover ---------------------
__global__ void __launch_bounds__(256) mut2_kernel(
    const __bf16* __restrict__ mid, const __bf16* __restrict__ w2b,
    const float* __restrict__ b2, const float* __restrict__ pop,
    const int* __restrict__ idx, const unsigned char* __restrict__ mask,
    float* __restrict__ out) {
    extern __shared__ char smem[];
    __bf16* aTile = (__bf16*)smem;  // 16*2048 bf16
    const int tid = threadIdx.x;
    const int mt  = blockIdx.x;

    for (int e = tid; e < 16 * HDIM; e += 256)
        aTile[e] = mid[(size_t)mt * 16 * HDIM + e];
    __syncthreads();

    const int wave = tid >> 5, lane = tid & 31;
    const int n0   = (blockIdx.y * 8 + wave) * 16;
    v8f acc = {0.f, 0.f, 0.f, 0.f, 0.f, 0.f, 0.f, 0.f};
#pragma unroll 4
    for (int k0 = 0; k0 < HDIM; k0 += 32) {
        v16bf a = frag_a_lds(aTile, k0, lane);
        v16bf b = frag_b_gbl(w2b, n0, k0, lane);
        acc = __builtin_amdgcn_wmma_f32_16x16x32_bf16(
            false, a, false, b, (short)0, acc, false, false);
    }
    const int   n  = n0 + (lane & 15);
    const float bv = b2[n];
    const int   rb = (lane >> 4) * 8;
#pragma unroll
    for (int r = 0; r < 8; ++r) {
        int grow = mt * 16 + rb + r;
        if (grow < KSEL) {
            float mval  = acc[r] + bv;
            bool  take  = mask[(size_t)grow * HDIM + n] != 0;
            float bestv = pop[(size_t)idx[grow] * HDIM + n];
            out[(size_t)grow * HDIM + n] = take ? bestv : mval;
        }
    }
}

// ------------------------------ launch -------------------------------------
extern "C" void kernel_launch(void* const* d_in, const int* in_sizes, int n_in,
                              void* d_out, int out_size, void* d_ws, size_t ws_size,
                              hipStream_t stream) {
    const float* pop  = (const float*)d_in[0];
    const float* sw1  = (const float*)d_in[1];
    const float* sb1  = (const float*)d_in[2];
    const float* sw2  = (const float*)d_in[3];
    const float* sb2  = (const float*)d_in[4];
    const float* mw1  = (const float*)d_in[5];
    const float* mb1  = (const float*)d_in[6];
    const float* mw2  = (const float*)d_in[7];
    const float* mb2  = (const float*)d_in[8];
    const unsigned char* mask = (const unsigned char*)d_in[9];

    float* out = (float*)d_out;                 // [0, 50*2048): offspring
    float* fit_out = out + (size_t)KSEL * HDIM; // [50*2048, +65536): fitness

    char* ws = (char*)d_ws;
    __bf16* w1b   = (__bf16*)(ws + WS_SELW1B);
    __bf16* mw1b  = (__bf16*)(ws + WS_MUTW1B);
    __bf16* mw2b  = (__bf16*)(ws + WS_MUTW2B);
    float*  fcopy = (float*)(ws + WS_FITCOPY);
    int*    idx   = (int*)(ws + WS_IDX);
    __bf16* mid   = (__bf16*)(ws + WS_MID);

    const size_t ldsA16 = (size_t)16 * HDIM * 2;  //  64KB bf16 A tile
    const size_t ldsA32 = (size_t)32 * HDIM * 2;  // 128KB (two A tiles)

    cvt_transpose_bf16<<<2048, 256, 0, stream>>>(sw1, w1b, HDIM, HHALF);
    cvt_transpose_bf16<<<4096, 256, 0, stream>>>(mw1, mw1b, HDIM, HDIM);
    cvt_transpose_bf16<<<4096, 256, 0, stream>>>(mw2, mw2b, HDIM, HDIM);

    fitness_kernel<<<N_POP / 32, 256, ldsA32 + 128, stream>>>(
        pop, w1b, sb1, sw2, sb2, fit_out, fcopy);

    topk_kernel<<<1, 1024, 0, stream>>>(fcopy, idx);

    mut1_kernel<<<dim3(4, 16), 256, ldsA16 + 64, stream>>>(pop, idx, mw1b, mb1, mid);
    mut2_kernel<<<dim3(4, 16), 256, ldsA16, stream>>>(mid, mw2b, mb2, pop, idx, mask, out);
}
